// AsMac_33311766348306
// MI455X (gfx1250) — matrix operand order repaired and checked by reading, not compiled, not run on hardware
//
#include <hip/hip_runtime.h>
#include <hip/hip_bf16.h>

// ---------------- problem constants (from reference) ----------------
#define GAMMA_C 0.01f
#define INVG    100.0f      // 1/GAMMA
#define KK      64          // motif length
#define BB      16          // batch
#define DD      5           // alphabet
#define LL      1024        // sequence length
#define CC      256         // channels
#define NJ      (LL - KK + 1)   // 961 valid window starts

typedef __attribute__((ext_vector_type(16))) _Float16 v16h;
typedef __attribute__((ext_vector_type(8)))  _Float16 v8h;
typedef __attribute__((ext_vector_type(8)))  float    v8f;

__device__ __forceinline__ unsigned int f2ord(float f) {
    // map float -> monotonically ordered uint
    unsigned int u = __float_as_uint(f);
    return (u & 0x80000000u) ? ~u : (u | 0x80000000u);
}

// ---------------------------------------------------------------
// Kernel 1: prep — weights f32->f16, tokens = argmax_d(one-hot),
//           zero the packed argmax accumulators (re-done every launch).
// ---------------------------------------------------------------
__global__ void k_prep(const float* __restrict__ seq,
                       const float* __restrict__ w,
                       _Float16* __restrict__ Wh,
                       int* __restrict__ tok,
                       unsigned long long* __restrict__ packed) {
    int idx = blockIdx.x * blockDim.x + threadIdx.x;
    if (idx < CC * DD * KK) Wh[idx] = (_Float16)w[idx];       // 81920
    if (idx < BB * LL) {                                      // 16384
        int b = idx >> 10, l = idx & 1023;
        const float* p = seq + b * DD * LL + l;
        int best = 0; float bv = p[0];
        #pragma unroll
        for (int d = 1; d < DD; ++d) {
            float v = p[d * LL];
            if (v > bv) { bv = v; best = d; }
        }
        tok[idx] = best;
    }
    if (idx < BB * CC) packed[idx] = 0ull;                    // 4096
}

// ---------------------------------------------------------------
// Kernel 2: WMMA conv (im2col GEMM) + packed-u64 atomic argmax.
// Grid: (16 j-tiles of 64, BB).  Block: 256 = 8 waves.
// Wave owns 2 c-tiles (16 rows each); per c-tile it loads ALL 10 A
// fragments (K=320) into registers once, then sweeps 4 j-sub-tiles
// with pure LDS-gather + back-to-back WMMA (no global waits inside).
// ---------------------------------------------------------------
__global__ void k_conv_argmax(const float* __restrict__ seq,
                              const _Float16* __restrict__ Wh,
                              unsigned long long* __restrict__ packed) {
    __shared__ _Float16 sld[DD * 128];   // seq window [d][128] in f16
    const int b   = blockIdx.y;
    const int j0  = blockIdx.x * 64;
    const int tid = threadIdx.x;

    // stage sliding seq window (j0 .. j0+126) to LDS, converting to f16
    for (int idx = tid; idx < DD * 128; idx += 256) {
        int d = idx >> 7, t = idx & 127;
        int l = j0 + t;
        float v = (l < LL) ? seq[(b * DD + d) * LL + l] : 0.0f;
        sld[idx] = (_Float16)v;
    }
    __syncthreads();

    const int wid  = tid >> 5;
    const int lane = tid & 31;
    const int hi   = lane >> 4;    // lane half-group (0: 0-15, 1: 16-31)
    const int ln16 = lane & 15;

    #pragma unroll 1
    for (int ct = 0; ct < 2; ++ct) {
        const int c0 = (wid + 8 * ct) * 16;              // channel tile base
        const _Float16* wrow = Wh + (c0 + ln16) * 320;   // A row (M = ln16)
        __builtin_prefetch(wrow + 8 * 16 * 320, 0, 1);   // next c-tile's rows

        // ---- load all 10 A fragments for this c-tile into registers.
        // 16x32 f16 A layout: lanes<16 hold K {0..7,16..23},
        // lanes>=16 hold K {8..15,24..31} relative to the 32-chunk base.
        v16h afrag[10];
        #pragma unroll
        for (int kc = 0; kc < 10; ++kc) {
            const int b1 = kc * 32 + hi * 8;
            v8h alo = *(const v8h*)(wrow + b1);      // 16B aligned (b1 % 8 == 0)
            v8h ahi = *(const v8h*)(wrow + b1 + 16);
            #pragma unroll
            for (int q = 0; q < 8; ++q) { afrag[kc][q] = alo[q]; afrag[kc][q + 8] = ahi[q]; }
        }

        // ---- sweep the 4 j-sub-tiles; A stays resident in VGPRs.
        #pragma unroll 1
        for (int js = 0; js < 4; ++js) {
            const int jsub = js * 16;
            v8f acc = {};
            #pragma unroll
            for (int kc = 0; kc < 10; ++kc) {
                // B fragment: 32x16 f16.  lanes<16 hold K {0..7,16..23},
                // lanes>=16 hold K {8..15,24..31}; N = lane%16.
                const int kb = kc * 32 + hi * 8;
                const int d  = kb >> 6;        // alphabet plane
                const int ks = kb & 63;        // offset inside motif
                const _Float16* p = &sld[d * 128 + jsub + ln16 + ks];
                v16h bfrag;
                #pragma unroll
                for (int q = 0; q < 16; ++q) bfrag[q] = p[q < 8 ? q : q + 8];
                acc = __builtin_amdgcn_wmma_f32_16x16x32_f16(
                        false, afrag[kc], false, bfrag, (short)0, acc, false, false);
            }
            // ---- packed argmax over j (first-max tie-break like jnp.argmax)
            const int j = j0 + jsub + ln16;
            if (j < NJ) {
                #pragma unroll
                for (int r = 0; r < 8; ++r) {
                    int c = c0 + hi * 8 + r;   // C/D layout: M = r + 8*(lane/16)
                    unsigned long long pk =
                        ((unsigned long long)f2ord(acc[r]) << 32) |
                        (unsigned long long)(0xFFFFFFFFu - (unsigned)j);
                    atomicMax(&packed[b * CC + c], pk);
                }
            }
        }
    }
}

// ---------------------------------------------------------------
// Kernel 3: soft Needleman-Wunsch, one wave32 per (b,c) pair.
// 64x64 DP, anti-diagonal wavefront, lane = column, two 32-col passes.
// s[i][j] = W[c][tok[jm+i]][j]  (one-hot einsum collapses to gather).
// ---------------------------------------------------------------
__global__ void k_softnw(const float* __restrict__ weights,
                         const float* __restrict__ gap,
                         const int* __restrict__ tok,
                         const unsigned long long* __restrict__ packed,
                         float* __restrict__ scores) {
    __shared__ float colbuf[8][65];   // D[i][32] boundary between passes
    __shared__ int   tokb[8][KK];
    const int tid  = threadIdx.x;
    const int wid  = tid >> 5;
    const int lane = tid & 31;
    const int pair = blockIdx.x * 8 + wid;    // pair = b*CC + c
    const int b = pair >> 8;
    const int c = pair & 255;

    const float g = gap[c];
    const unsigned long long P = packed[pair];
    const int jm = (int)(0xFFFFFFFFu - (unsigned)(P & 0xFFFFFFFFu));

    for (int i = lane; i < KK; i += 32) tokb[wid][i] = tok[b * LL + jm + i];
    if (lane == 0) colbuf[wid][0] = -32.0f * g;    // D[0][32]
    __syncthreads();

    float finalv = 0.0f;
    #pragma unroll 1
    for (int pass = 0; pass < 2; ++pass) {
        const int jj = pass * 32 + lane;                 // s-column 0..63
        const float* wp = weights + c * (DD * KK) + jj;  // per-lane W column
        const float w0 = wp[0 * KK], w1 = wp[1 * KK], w2 = wp[2 * KK],
                    w3 = wp[3 * KK], w4 = wp[4 * KK];
        float cur = -(float)(jj + 1) * g;                // D[0][jj+1]
        float leftPrev = 0.0f;                           // D[0][jj] for lane 0
        if (lane == 0) leftPrev = (pass == 0) ? 0.0f : colbuf[wid][0];

        for (int t = 0; t < KK + 31; ++t) {
            float leftN = __shfl_up(cur, 1, 32);         // D[i][jj] from lane-1
            if (lane == 0) {
                int ii = t + 1;
                leftN = (pass == 0) ? (-(float)ii * g)
                                    : colbuf[wid][ii <= KK ? ii : KK];
            }
            const int i = t - lane + 1;                  // DP row (1-based)
            if (i >= 1 && i <= KK) {
                const int tk = tokb[wid][i - 1];
                float sij = w0;
                sij = (tk == 1) ? w1 : sij;
                sij = (tk == 2) ? w2 : sij;
                sij = (tk == 3) ? w3 : sij;
                sij = (tk == 4) ? w4 : sij;
                const float a0 = leftPrev;               // diag D[i-1][jj]
                const float a1 = cur - g;                // up   D[i-1][jj+1] - g
                const float a2 = leftN - g;              // left D[i][jj]     - g
                const float m  = fmaxf(a0, fmaxf(a1, a2));
                const float ss = __expf((a0 - m) * INVG) +
                                 __expf((a1 - m) * INVG) +
                                 __expf((a2 - m) * INVG);
                const float val = sij + m + GAMMA_C * __logf(ss);
                cur = val;
                if (pass == 0 && lane == 31) colbuf[wid][i] = val;  // D[i][32]
            }
            leftPrev = leftN;
        }
        __syncthreads();
        finalv = cur;
    }
    if (lane == 31) scores[pair] = finalv;   // D[64][64]
}

// ---------------------------------------------------------------
// Kernel 4: relu(scores+bias), L2-normalize rows, 1 - Gram matrix.
// ---------------------------------------------------------------
__global__ void k_finalize(const float* __restrict__ scores,
                           const float* __restrict__ bias,
                           float* __restrict__ out) {
    __shared__ float e[BB][CC];
    __shared__ float nrm[BB];
    const int tid = threadIdx.x;
    for (int idx = tid; idx < BB * CC; idx += 256) {
        int c = idx & 255;
        float v = scores[idx] + bias[c];
        e[idx >> 8][c] = v > 0.0f ? v : 0.0f;
    }
    __syncthreads();
    if (tid < BB) {
        float s = 0.0f;
        for (int c = 0; c < CC; ++c) { float v = e[tid][c]; s += v * v; }
        nrm[tid] = fmaxf(sqrtf(s), 1e-12f);
    }
    __syncthreads();
    if (tid < BB * BB) {
        const int a = tid >> 4, b2 = tid & 15;
        float dot = 0.0f;
        for (int c = 0; c < CC; ++c) dot += e[a][c] * e[b2][c];
        out[tid] = 1.0f - dot / (nrm[a] * nrm[b2]);
    }
}

// ---------------------------------------------------------------
extern "C" void kernel_launch(void* const* d_in, const int* in_sizes, int n_in,
                              void* d_out, int out_size, void* d_ws, size_t ws_size,
                              hipStream_t stream) {
    (void)in_sizes; (void)n_in; (void)out_size; (void)ws_size;
    const float* seq  = (const float*)d_in[0];   // [16,5,1024]
    const float* w    = (const float*)d_in[1];   // [256,5,64]
    const float* bias = (const float*)d_in[2];   // [256]
    const float* gap  = (const float*)d_in[3];   // [256]
    float* out = (float*)d_out;                  // [16,16]

    char* ws = (char*)d_ws;                                  // layout (16B aligned):
    _Float16*           Wh     = (_Float16*)(ws);            // 163840 B
    int*                tok    = (int*)(ws + 163840);        //  65536 B
    unsigned long long* packed = (unsigned long long*)(ws + 229376); // 32768 B
    float*              scores = (float*)(ws + 262144);      //  16384 B

    k_prep       <<<320, 256, 0, stream>>>(seq, w, Wh, tok, packed);
    k_conv_argmax<<<dim3(16, BB), 256, 0, stream>>>(seq, Wh, packed);
    k_softnw     <<<(BB * CC) / 8, 256, 0, stream>>>(w, gap, tok, packed, scores);
    k_finalize   <<<1, 256, 0, stream>>>(scores, bias, out);
}
